// PhysicsAttention_12111807775270
// MI455X (gfx1250) — matrix operand order repaired and checked by reading, not compiled, not run on hardware
//
#include <hip/hip_runtime.h>

#define Nn 64
#define Cc 128
#define Tt 256
#define Vv 25
#define Oo 128

typedef __attribute__((ext_vector_type(16))) __bf16 v16bf;
typedef __attribute__((ext_vector_type(8)))  float  v8f;

union Frag16 { unsigned int u[8]; v16bf v; };
union FragC  { float f[8]; v8f v; };

// bf16 conversion, round-half-up (cheap: add + shift; pack = add,add,v_perm)
static __device__ __forceinline__ unsigned short f2bf(float f) {
  return (unsigned short)((__float_as_uint(f) + 0x8000u) >> 16);
}
static __device__ __forceinline__ unsigned int pack_bf16(float lo, float hi) {
  unsigned int a = __float_as_uint(lo) + 0x8000u;
  unsigned int b = __float_as_uint(hi) + 0x8000u;
  return __byte_perm(a, b, 0x7632);   // {b[3],b[2],a[3],a[2]} -> one v_perm_b32
}

// A-matrix (16-bit, 16x32) K index for VGPR j at this lane (even; pair = k, k+1)
static __device__ __forceinline__ int a_kmap(int j, int lane) {
  return ((j < 4) ? (2 * j) : (16 + 2 * (j - 4))) + ((lane >> 4) << 3);
}
// B-matrix (16-bit, 32x16) K index for VGPR j at this lane (even; pair = k, k+1)
static __device__ __forceinline__ int b_kmap(int j, int lane) {
  return ((j < 4) ? (2 * j) : (16 + 2 * (j - 4))) + ((lane >> 4) << 3);
}

// Pre-swizzle the three 128x128 f32 weights into bf16 WMMA A-fragments.
// Layout: wfrag[(p*32 + ot*4 + ks)*256 + j*32 + lane]  (1 dword = bf16 pair)
__global__ void __launch_bounds__(32)
prep_weights(const float* __restrict__ wq, const float* __restrict__ wk,
             const float* __restrict__ wv, unsigned int* __restrict__ wfrag) {
  const int lane = threadIdx.x;
  const int f = blockIdx.x & 31;   // ot*4 + ks
  const int p = blockIdx.x >> 5;   // 0=q 1=k 2=v
  const float* w = (p == 0) ? wq : (p == 1) ? wk : wv;
  const int ot = f >> 2, ks = f & 3;
  const int row = ot * 16 + (lane & 15);
#pragma unroll
  for (int j = 0; j < 8; ++j) {
    int k = ks * 32 + a_kmap(j, lane);
    wfrag[(p * 32 + f) * 256 + j * 32 + lane] =
        pack_bf16(w[row * Cc + k], w[row * Cc + k + 1]);
  }
}

// Materialize the 25x25 physics-bias matrix once (constant across problems).
__global__ void __launch_bounds__(32)
prep_bias(const float* __restrict__ bias_emb, const int* __restrict__ hop,
          float* __restrict__ bias_f) {
  for (int i = threadIdx.x; i < Vv * Vv; i += 32)
    bias_f[i] = bias_emb[hop[i]];
}

// One wave32 per (n,t) attention problem, all matmuls via v_wmma_f32_16x16x32_bf16.
__global__ void __launch_bounds__(32)
physics_attn(const float* __restrict__ x,
             const float* __restrict__ bq, const float* __restrict__ bk,
             const float* __restrict__ bv,
             const float* __restrict__ bias_f,
             const unsigned int* __restrict__ wfrag,
             float* __restrict__ out) {
  // lds_x dies after the projections; scores/probs alias onto it.
  __shared__ union {
    unsigned short x[Cc * 32];                 // X^T pair-packed by channel
    struct {
      float          s[32 * 33];               // f32 scores, padded stride
      unsigned short p[32 * 32];               // bf16 probs, row-major [v][32]
    } sp;
  } ubuf;
  __shared__ unsigned short lds_q[Oo * 32];    // pair-packed by o
  __shared__ unsigned short lds_k[Oo * 32];    // pair-packed by o
  __shared__ unsigned short lds_v[Oo * 32];    // row-major [o][32]
  __shared__ float          sbias[3 * Cc];

  const int lane = threadIdx.x;
  const int h8   = (lane >> 4) << 3;           // 0 or 8
  const int lo16 = lane & 15;
  const int prob = blockIdx.x;
  const int n = prob / Tt, t = prob % Tt;
  const size_t xbase = (size_t)n * (Cc * Tt * Vv) + (size_t)t * Vv;

  // ---- stage biases ----
  for (int i = lane; i < Cc; i += 32) {
    sbias[i]          = bq[i];
    sbias[Cc + i]     = bk[i];
    sbias[2 * Cc + i] = bv[i];
  }

  // ---- load X^T (C x V) -> LDS bf16, pair-packed by channel; pad V to 32 ----
  for (int i = lane; i < Cc * Vv; i += 32) {
    int c = i / Vv, v = i - c * Vv;
    float val = x[xbase + (size_t)c * (Tt * Vv) + v];
    ubuf.x[(c >> 1) * 64 + v * 2 + (c & 1)] = f2bf(val);
  }
  {  // zero-pad columns v=25..31 (whole channel-pair dwords)
    unsigned int* lxw = (unsigned int*)ubuf.x;
#pragma unroll
    for (int v = Vv; v < 32; ++v)
      for (int cp = lane; cp < Cc / 2; cp += 32) lxw[cp * 32 + v] = 0u;
  }
  __syncthreads();

  // ---- hoist all X^T B-fragments into registers (shared by q,k,v & both vt) --
  const unsigned int* lx = (const unsigned int*)ubuf.x;
  Frag16 bx[2][4];
#pragma unroll
  for (int vt = 0; vt < 2; ++vt)
#pragma unroll
    for (int ks = 0; ks < 4; ++ks)
#pragma unroll
      for (int j = 0; j < 8; ++j) {
        int k = ks * 32 + b_kmap(j, lane);
        bx[vt][ks].u[j] = lx[(k >> 1) * 32 + vt * 16 + lo16];
      }

  // ---- Q,K projections (pair-packed store, packed b32 writes) ----
  for (int p = 0; p < 2; ++p) {
    unsigned int* dst = (unsigned int*)(p ? lds_k : lds_q);
    const unsigned int* wf = wfrag + p * 32 * 256;
    for (int ot = 0; ot < 8; ++ot) {
      Frag16 a[4];
#pragma unroll
      for (int ks = 0; ks < 4; ++ks)
#pragma unroll
        for (int j = 0; j < 8; ++j)
          a[ks].u[j] = wf[(ot * 4 + ks) * 256 + j * 32 + lane];
      float bias8[8];
#pragma unroll
      for (int r = 0; r < 8; ++r) bias8[r] = sbias[p * Cc + ot * 16 + r + h8];
#pragma unroll
      for (int vt = 0; vt < 2; ++vt) {
        FragC acc;
#pragma unroll
        for (int r = 0; r < 8; ++r) acc.f[r] = bias8[r];
#pragma unroll
        for (int ks = 0; ks < 4; ++ks)
          acc.v = __builtin_amdgcn_wmma_f32_16x16x32_bf16(
              false, a[ks].v, false, bx[vt][ks].v, (short)0, acc.v, false, false);
        int vv = vt * 16 + lo16;
#pragma unroll
        for (int r = 0; r < 8; r += 2)
          dst[(ot * 8 + ((r + h8) >> 1)) * 32 + vv] =
              pack_bf16(acc.f[r], acc.f[r + 1]);
      }
    }
  }

  // ---- V projection (row-major store) ----
  {
    const unsigned int* wf = wfrag + 2 * 32 * 256;
    for (int ot = 0; ot < 8; ++ot) {
      Frag16 a[4];
#pragma unroll
      for (int ks = 0; ks < 4; ++ks)
#pragma unroll
        for (int j = 0; j < 8; ++j)
          a[ks].u[j] = wf[(ot * 4 + ks) * 256 + j * 32 + lane];
      float bias8[8];
#pragma unroll
      for (int r = 0; r < 8; ++r) bias8[r] = sbias[2 * Cc + ot * 16 + r + h8];
#pragma unroll
      for (int vt = 0; vt < 2; ++vt) {
        FragC acc;
#pragma unroll
        for (int r = 0; r < 8; ++r) acc.f[r] = bias8[r];
#pragma unroll
        for (int ks = 0; ks < 4; ++ks)
          acc.v = __builtin_amdgcn_wmma_f32_16x16x32_bf16(
              false, a[ks].v, false, bx[vt][ks].v, (short)0, acc.v, false, false);
        int vv = vt * 16 + lo16;
#pragma unroll
        for (int r = 0; r < 8; ++r)
          lds_v[(ot * 16 + r + h8) * 32 + vv] = f2bf(acc.f[r]);
      }
    }
  }
  __syncthreads();

  // ---- scores: S = (Q^T)^T * K^T  (fragments hoisted once) ----
  {
    const unsigned int* lq = (const unsigned int*)lds_q;
    const unsigned int* lk = (const unsigned int*)lds_k;
    Frag16 aq[2][4], bk_[2][4];
#pragma unroll
    for (int mt = 0; mt < 2; ++mt)
#pragma unroll
      for (int ks = 0; ks < 4; ++ks)
#pragma unroll
        for (int j = 0; j < 8; ++j) {
          int ko = ks * 32 + a_kmap(j, lane);
          aq[mt][ks].u[j] = lq[(ko >> 1) * 32 + mt * 16 + lo16];
        }
#pragma unroll
    for (int nt = 0; nt < 2; ++nt)
#pragma unroll
      for (int ks = 0; ks < 4; ++ks)
#pragma unroll
        for (int j = 0; j < 8; ++j) {
          int ko = ks * 32 + b_kmap(j, lane);
          bk_[nt][ks].u[j] = lk[(ko >> 1) * 32 + nt * 16 + lo16];
        }
#pragma unroll
    for (int mt = 0; mt < 2; ++mt)
#pragma unroll
      for (int nt = 0; nt < 2; ++nt) {
        FragC acc;
#pragma unroll
        for (int r = 0; r < 8; ++r) acc.f[r] = 0.0f;
#pragma unroll
        for (int ks = 0; ks < 4; ++ks)
          acc.v = __builtin_amdgcn_wmma_f32_16x16x32_bf16(
              false, aq[mt][ks].v, false, bk_[nt][ks].v, (short)0, acc.v,
              false, false);
#pragma unroll
        for (int r = 0; r < 8; ++r)
          ubuf.sp.s[(mt * 16 + r + h8) * 33 + nt * 16 + lo16] = acc.f[r];
      }
  }
  __syncthreads();

  // ---- softmax over w (one lane per row v), precomputed bias matrix ----
  {
    const float inv_sqrt = 0.08838834764831845f;  // 1/sqrt(128)
    unsigned int* pw = (unsigned int*)ubuf.sp.p;
    if (lane < Vv) {
      float vals[Vv];
      float mx = -3.0e38f;
#pragma unroll
      for (int w = 0; w < Vv; ++w) {
        float s = ubuf.sp.s[lane * 33 + w] * inv_sqrt + bias_f[lane * Vv + w];
        vals[w] = s;
        mx = fmaxf(mx, s);
      }
      float sum = 0.0f;
#pragma unroll
      for (int w = 0; w < Vv; ++w) {
        float e = __expf(vals[w] - mx);
        vals[w] = e;
        sum += e;
      }
      float inv = 1.0f / sum;
#pragma unroll
      for (int w2 = 0; w2 < 16; ++w2) {
        float lo = (2 * w2 < Vv) ? vals[2 * w2] * inv : 0.0f;
        float hi = (2 * w2 + 1 < Vv) ? vals[2 * w2 + 1] * inv : 0.0f;
        pw[lane * 16 + w2] = pack_bf16(lo, hi);
      }
    } else {
#pragma unroll
      for (int w2 = 0; w2 < 16; ++w2) pw[lane * 16 + w2] = 0u;
    }
  }
  __syncthreads();

  // ---- output: Out = P * V  -> lanes hold 8 consecutive v, vector stores ----
  {
    const unsigned int* lv = (const unsigned int*)lds_v;
    const unsigned int* lp = (const unsigned int*)ubuf.sp.p;
    Frag16 ap[2];
#pragma unroll
    for (int mt = 0; mt < 2; ++mt)
#pragma unroll
      for (int j = 0; j < 8; ++j) {
        int kw = a_kmap(j, lane);              // even, 0..31
        ap[mt].u[j] = lp[(mt * 16 + lo16) * 16 + (kw >> 1)];
      }
    for (int ot = 0; ot < 8; ++ot) {
      Frag16 bv_;
#pragma unroll
      for (int j = 0; j < 8; ++j) {
        int kw = b_kmap(j, lane);              // even, 0..31
        bv_.u[j] = lv[(ot * 16 + lo16) * 16 + (kw >> 1)];
      }
      const int o = ot * 16 + lo16;
      float* op = out + (size_t)n * (Oo * Tt * Vv) + (size_t)o * (Tt * Vv) +
                  (size_t)t * Vv;
#pragma unroll
      for (int mt = 0; mt < 2; ++mt) {
        FragC acc;
#pragma unroll
        for (int r = 0; r < 8; ++r) acc.f[r] = 0.0f;
        acc.v = __builtin_amdgcn_wmma_f32_16x16x32_bf16(
            false, ap[mt].v, false, bv_.v, (short)0, acc.v, false, false);
        const int v0 = mt * 16 + h8;           // first of 8 consecutive v
        if (mt == 0) {                         // v0 in {0,8}: all valid
          float4 s0 = make_float4(acc.f[0], acc.f[1], acc.f[2], acc.f[3]);
          float4 s1 = make_float4(acc.f[4], acc.f[5], acc.f[6], acc.f[7]);
          *(float4*)(op + v0)     = s0;
          *(float4*)(op + v0 + 4) = s1;
        } else {
          if (lane < 16) {                     // v0 = 16: v 16..23 valid
            float4 s0 = make_float4(acc.f[0], acc.f[1], acc.f[2], acc.f[3]);
            float4 s1 = make_float4(acc.f[4], acc.f[5], acc.f[6], acc.f[7]);
            *(float4*)(op + 16) = s0;
            *(float4*)(op + 20) = s1;
          } else {                             // v0 = 24: only v=24 valid
            op[24] = acc.f[0];
          }
        }
      }
    }
  }
}

extern "C" void kernel_launch(void* const* d_in, const int* in_sizes, int n_in,
                              void* d_out, int out_size, void* d_ws, size_t ws_size,
                              hipStream_t stream) {
  const float* x        = (const float*)d_in[0];
  const float* wq       = (const float*)d_in[1];
  const float* bq       = (const float*)d_in[2];
  const float* wk       = (const float*)d_in[3];
  const float* bk       = (const float*)d_in[4];
  const float* wv       = (const float*)d_in[5];
  const float* bv       = (const float*)d_in[6];
  const float* bias_emb = (const float*)d_in[7];
  const int*   hop      = (const int*)d_in[8];

  unsigned int* wfrag = (unsigned int*)d_ws;          // 3*32*256 dwords = 96 KB
  float* bias_f = (float*)((char*)d_ws + 3 * 32 * 256 * 4);  // 25*25 floats

  prep_weights<<<96, 32, 0, stream>>>(wq, wk, wv, wfrag);
  prep_bias<<<1, 32, 0, stream>>>(bias_emb, hop, bias_f);
  physics_attn<<<Nn * Tt, 32, 0, stream>>>(x, bq, bk, bv, bias_f,
                                           wfrag, (float*)d_out);
}
